// CAPMemory_33148557591294
// MI455X (gfx1250) — matrix operand order repaired and checked by reading, not compiled, not run on hardware
//
#include <hip/hip_runtime.h>
#include <hip/hip_bf16.h>

typedef __attribute__((ext_vector_type(16))) _Float16 v16h;
typedef __attribute__((ext_vector_type(8)))  float    v8f;
typedef __attribute__((ext_vector_type(4)))  _Float16 h4;

#define NSAMP 512
#define DDIM  256
#define LLAB  8000
#define CCAM  8
#define PPROX 64000          // L*C
#define TILES 4000           // P/16
#define BANDS 32             // N/16
#define CHUNKS 16
#define TPC   250            // TILES/CHUNKS
#define NBLK  (BANDS * CHUNKS)   // 512 blocks
#define NEGINF (-3.0e38f)
#define INVT  20.0f          // 1/T

// ---- workspace layout (bytes) ----
static constexpr size_t OFF_CENH  = 0;                       // 64000*256*2
static constexpr size_t OFF_FEATH = OFF_CENH  + (size_t)PPROX * DDIM * 2;
static constexpr size_t OFF_LAB   = OFF_FEATH + (size_t)NSAMP * DDIM * 2;
static constexpr size_t OFF_CAM   = OFF_LAB   + NSAMP * 4;
static constexpr size_t OFF_TI    = OFF_CAM   + NSAMP * 4;
static constexpr size_t OFF_POS   = OFF_TI    + NSAMP * 4;        // 512*8 floats
static constexpr size_t OFF_PM    = OFF_POS   + NSAMP * CCAM * 4; // NBLK*16 floats
static constexpr size_t OFF_PS    = OFF_PM    + (size_t)NBLK * 16 * 4;
static constexpr size_t OFF_TOP   = OFF_PS    + (size_t)NBLK * 16 * 4;  // NBLK*16*50
static constexpr size_t OFF_LIN   = OFF_TOP   + (size_t)NBLK * 16 * 50 * 4;
static constexpr size_t OFF_LIT   = OFF_LIN   + NSAMP * 4;
static constexpr size_t WS_NEEDED = OFF_LIT   + NSAMP * 4;

// =============== kernel 1: normalize feats -> f16, gather labels/cams ===========
__global__ void __launch_bounds__(256) prep_kernel(
    const float* __restrict__ feats, const int* __restrict__ indexes,
    const int* __restrict__ ltab, const int* __restrict__ ctab,
    _Float16* __restrict__ featsH, int* __restrict__ labs, int* __restrict__ cams)
{
  __shared__ float red[256];
  const int b = blockIdx.x, t = threadIdx.x;
  float v = feats[b * DDIM + t];
  red[t] = v * v;
  __syncthreads();
  for (int s = 128; s > 0; s >>= 1) {
    if (t < s) red[t] += red[t + s];
    __syncthreads();
  }
  float inv = rsqrtf(red[0]);
  featsH[b * DDIM + t] = (_Float16)(v * inv);
  if (t == 0) {
    int idx = indexes[b];
    labs[b] = ltab[idx];
    cams[b] = ctab[idx];
  }
}

// =============== kernel 2: centers f32 -> f16 (one time) =======================
__global__ void __launch_bounds__(256) cvt_centers(
    const float* __restrict__ cen, _Float16* __restrict__ cenH, int nquad)
{
  int i = blockIdx.x * blockDim.x + threadIdx.x;
  const int stride = gridDim.x * blockDim.x;
  for (; i < nquad; i += stride) {
    float4 f = ((const float4*)cen)[i];
    h4 h; h[0] = (_Float16)f.x; h[1] = (_Float16)f.y; h[2] = (_Float16)f.z; h[3] = (_Float16)f.w;
    ((h4*)cenH)[i] = h;
  }
}

// wave-cooperative: sort-select top `keep` of buf[0..cnt) in-place (descending),
// update cnt/thr. Whole wave must call together.
__device__ __forceinline__ void compact_row(float* bufp, int* cntp, float* thrp,
                                            int lane, int keep)
{
  volatile float* buf = (volatile float*)bufp;
  volatile int*   cnt = (volatile int*)cntp;
  volatile float* thr = (volatile float*)thrp;
  int n = *cnt; if (n > 80) n = 80;
  float v0 = (lane      < n) ? buf[lane]      : NEGINF;
  float v1 = (lane + 32 < n) ? buf[lane + 32] : NEGINF;
  float v2 = (lane + 64 < n) ? buf[lane + 64] : NEGINF;
  const int m = (n < keep) ? n : keep;
  for (int i = 0; i < m; ++i) {
    float wmax = fmaxf(v0, fmaxf(v1, v2));
#pragma unroll
    for (int d = 16; d > 0; d >>= 1) wmax = fmaxf(wmax, __shfl_xor(wmax, d, 32));
    bool have = (v0 == wmax) | (v1 == wmax) | (v2 == wmax);
    unsigned long long bm = __ballot(have);
    int owner = __ffsll(bm) - 1;
    if (lane == owner) {
      if (v0 == wmax) v0 = NEGINF;
      else if (v1 == wmax) v1 = NEGINF;
      else v2 = NEGINF;
    }
    if (lane == 0) buf[i] = wmax;   // sources already in registers
  }
  if (lane == 0) { *cnt = m; *thr = (m > 0) ? buf[m - 1] : NEGINF; }
  __builtin_amdgcn_wave_barrier();
}

// =============== kernel 3: fused f16-WMMA GEMM + streaming statistics ==========
__global__ void __launch_bounds__(256) gemm_stats_kernel(
    const _Float16* __restrict__ featsH, const _Float16* __restrict__ cenH,
    const int* __restrict__ labs, const int* __restrict__ cams,
    float* __restrict__ posG, float* __restrict__ tiG,
    float* __restrict__ partM, float* __restrict__ partS, float* __restrict__ partTop)
{
  __shared__ int   s_lab[16], s_cam[16];
  __shared__ float s_buf[8][16][80];   // per-wave negative candidate buffers
  __shared__ int   s_cnt[8][16];
  __shared__ float s_thr[8][16];
  __shared__ float s_pm[8][16][2];     // intra-LSE partial maxes (2 lanes/row/wave)
  __shared__ float s_ps[8][16][2];     // intra-LSE partial sums

  const int tid  = threadIdx.x;
  const int w    = tid >> 5;
  const int lane = tid & 31;
  const int band  = blockIdx.x & (BANDS - 1);  // sample row-band (16 samples)
  const int chunk = blockIdx.x / BANDS;        // proxy chunk (TPC column tiles)
  const int blk   = band * CHUNKS + chunk;

  if (tid < 16) {
    s_lab[tid] = labs[band * 16 + tid];
    s_cam[tid] = cams[band * 16 + tid];
  }
  if (tid < 128) {
    ((int*)s_cnt)[tid]   = 0;
    ((float*)s_thr)[tid] = NEGINF;
  }
  __syncthreads();

  // per-lane tile geometry (WMMA 16x16 f32 C layout: VGPR g = row, lane%16 = col)
  const int rbase = (lane >> 4) << 3;  // rows rbase..rbase+7 live in this lane
  int labr[8], camr[8];
#pragma unroll
  for (int g = 0; g < 8; ++g) { labr[g] = s_lab[rbase + g]; camr[g] = s_cam[rbase + g]; }

  // A fragments (16x32 f16 layout), loaded once and reused over all TPC tiles
  union V16 { v16h v; uint4 q[2]; };
  V16 afrag[8];
  {
    const int aM = lane & 15;
    const int kb = (lane >> 4) << 3;   // K sub-base 0 or 8
    const _Float16* arow = featsH + (size_t)(band * 16 + aM) * DDIM;
#pragma unroll
    for (int kc = 0; kc < 8; ++kc) {
      afrag[kc].q[0] = *(const uint4*)(arow + kc * 32 + kb);
      afrag[kc].q[1] = *(const uint4*)(arow + kc * 32 + kb + 16);
    }
  }

  float mI[8], sI[8], thrReg[8];
#pragma unroll
  for (int g = 0; g < 8; ++g) { mI[g] = NEGINF; sI[g] = 0.f; thrReg[g] = NEGINF; }

  const int n16   = lane & 15;
  const int myCam = n16 & 7;
  const int lhalf = n16 >> 3;

  volatile int*   vcnt = (volatile int*)&s_cnt[w][0];
  volatile float* vthr = (volatile float*)&s_thr[w][0];

  const int tstart = chunk * TPC;
  for (int t = tstart + w; t < tstart + TPC; t += 8) {
    // ---- GEMM: 16x16 f32 tile via 8 x v_wmma_f32_16x16x32_f16 ----
    v8f acc = {0.f, 0.f, 0.f, 0.f, 0.f, 0.f, 0.f, 0.f};
    const _Float16* bcol = cenH + (size_t)(t * 16 + n16) * DDIM + ((lane >> 4) << 3);
    __builtin_prefetch(bcol + 8 * DDIM * 16, 0, 1);  // next tile for this wave
#pragma unroll
    for (int kc = 0; kc < 8; ++kc) {
      V16 bf;
      bf.q[0] = *(const uint4*)(bcol + kc * 32);
      bf.q[1] = *(const uint4*)(bcol + kc * 32 + 16);
      acc = __builtin_amdgcn_wmma_f32_16x16x32_f16(false, afrag[kc].v, false, bf.v,
                                                   (short)0, acc, false, false);
    }

    // ---- streaming statistics on the tile ----
    const int lIdx = 2 * t + lhalf;        // proxy label of this lane's column
#pragma unroll
    for (int g = 0; g < 8; ++g) {
      const float v   = acc[g];
      const int   row = rbase + g;
      const bool isPos    = (lIdx == labr[g]);
      const bool camMatch = (myCam == camr[g]);
      if (isPos) {                          // positives: written exactly once grid-wide
        const int gi = band * 16 + row;
        posG[gi * CCAM + myCam] = v;
        if (camMatch) tiG[gi] = v;
      }
      if (camMatch) {                       // intra-cam online LSE (includes label)
        if (v > mI[g]) { sI[g] = sI[g] * __expf((mI[g] - v) * INVT) + 1.0f; mI[g] = v; }
        else           { sI[g] += __expf((v - mI[g]) * INVT); }
      }
      if (!isPos && v > thrReg[g]) {        // negative candidate (reg-cached threshold)
        int idx = atomicAdd(&s_cnt[w][row], 1);
        ((volatile float*)&s_buf[w][row][0])[idx] = v;   // idx < 80 by construction
      }
    }
    // compact any row whose buffer could overflow next tile (<=16 appends/row/tile)
    bool need = (lane < 16) && (vcnt[lane] > 64);
    unsigned long long bm = __ballot(need);
    if (bm) {                                // wave-uniform branch
      do {
        int r = __ffsll(bm) - 1; bm &= bm - 1;
        compact_row(&s_buf[w][r][0], &s_cnt[w][r], &s_thr[w][r], lane, 50);
      } while (bm);
#pragma unroll
      for (int g = 0; g < 8; ++g) thrReg[g] = vthr[rbase + g];  // refresh reg cache
    }
  }

  // ---- per-wave finalization: exact sorted top-50 per row + intra partials ----
  for (int r = 0; r < 16; ++r) {
    compact_row(&s_buf[w][r][0], &s_cnt[w][r], &s_thr[w][r], lane, 50);
    int m = ((volatile int*)&s_cnt[w][r])[0];
    for (int j = lane; j < 50; j += 32)
      if (j >= m) ((volatile float*)&s_buf[w][r][0])[j] = NEGINF;
  }
#pragma unroll
  for (int g = 0; g < 8; ++g) {
    if (myCam == camr[g]) {
      const int slot = (lane >> 3) & 1;
      s_pm[w][rbase + g][slot] = mI[g];
      s_ps[w][rbase + g][slot] = sI[g];
    }
  }
  __syncthreads();

  // intra partial merge across 8 waves x 2 lanes -> one (M,S) per row
  if (tid < 16) {
    float M = NEGINF, S = 0.f;
    for (int c = 0; c < 8; ++c)
      for (int s2 = 0; s2 < 2; ++s2) {
        float m = s_pm[c][tid][s2], s = s_ps[c][tid][s2];
        if (m > M) { S = S * __expf((M - m) * INVT) + s; M = m; }
        else       { S += s * __expf((m - M) * INVT); }
      }
    partM[blk * 16 + tid] = M;
    partS[blk * 16 + tid] = S;
  }

  // merge 8 sorted per-wave top-50 lists -> block top-50 (lanes 0..7 hold heads)
  for (int r = w; r < 16; r += 8) {
    const int c = lane & 7;
    float head = (lane < 8) ? s_buf[c][r][0] : NEGINF;
    int   hi   = 0;
    float* out = partTop + ((size_t)blk * 16 + r) * 50;
    for (int k = 0; k < 50; ++k) {
      float wm = head;
#pragma unroll
      for (int d = 4; d > 0; d >>= 1) wm = fmaxf(wm, __shfl_xor(wm, d, 32));
      unsigned long long bmm = __ballot((lane < 8) && (head == wm));
      int owner = __ffsll(bmm) - 1;
      if (lane == owner) { ++hi; head = (hi < 50) ? s_buf[c][r][hi] : NEGINF; }
      if (lane == 0) out[k] = wm;
    }
  }
}

// =============== kernel 4: per-sample loss finalization ========================
__global__ void __launch_bounds__(256) finalize_kernel(
    const float* __restrict__ partM, const float* __restrict__ partS,
    const float* __restrict__ partTop, const float* __restrict__ posG,
    const float* __restrict__ tiG, float* __restrict__ lossIn, float* __restrict__ lossIt)
{
  const int i = blockIdx.x * blockDim.x + threadIdx.x;
  if (i >= NSAMP) return;
  const int band = i >> 4, row = i & 15;

  // intra: merge chunk LSE partials
  float M = NEGINF, S = 0.f;
  for (int c = 0; c < CHUNKS; ++c) {
    const int blk = band * CHUNKS + c;
    float m = partM[blk * 16 + row], s = partS[blk * 16 + row];
    if (m > M) { S = S * __expf((M - m) * INVT) + s; M = m; }
    else       { S += s * __expf((m - M) * INVT); }
  }
  lossIn[i] = M * INVT + __logf(S) - tiG[i] * INVT;

  // inter: CHUNKS-way merge of sorted top-50 lists + 8 positives
  const float* tp[CHUNKS]; float head[CHUNKS]; int hi[CHUNKS];
  float maxNeg = NEGINF;
  for (int c = 0; c < CHUNKS; ++c) {
    tp[c] = partTop + ((size_t)(band * CHUNKS + c) * 16 + row) * 50;
    head[c] = tp[c][0]; hi[c] = 0;
    maxNeg = fmaxf(maxNeg, head[c]);
  }
  float pv[CCAM], psum = 0.f, pmax = NEGINF;
  for (int c = 0; c < CCAM; ++c) {
    pv[c] = posG[i * CCAM + c];
    psum += pv[c];
    pmax = fmaxf(pmax, pv[c]);
  }
  const float M2 = fmaxf(maxNeg, pmax);
  float S2 = 0.f;
  for (int c = 0; c < CCAM; ++c) S2 += __expf((pv[c] - M2) * INVT);
  for (int k = 0; k < 50; ++k) {
    int best = 0; float bv = head[0];
#pragma unroll
    for (int c = 1; c < CHUNKS; ++c) if (head[c] > bv) { bv = head[c]; best = c; }
    S2 += __expf((bv - M2) * INVT);
    ++hi[best];
    head[best] = (hi[best] < 50) ? tp[best][hi[best]] : NEGINF;
  }
  lossIt[i] = M2 * INVT + __logf(S2) - psum * INVT * (1.0f / CCAM);
}

// =============== kernel 5: deterministic per-camera means -> out[2] ============
__global__ void __launch_bounds__(64) reduce_kernel(
    const float* __restrict__ lossIn, const float* __restrict__ lossIt,
    const int* __restrict__ cams, float* __restrict__ out)
{
  __shared__ float sA[CCAM], sB[CCAM];
  __shared__ int   sc[CCAM];
  const int t = threadIdx.x;
  if (t < CCAM) {
    float a = 0.f, b = 0.f; int cnt = 0;
    for (int i = 0; i < NSAMP; ++i)
      if (cams[i] == t) { a += lossIn[i]; b += lossIt[i]; ++cnt; }
    sA[t] = a; sB[t] = b; sc[t] = cnt;
  }
  __syncthreads();
  if (t == 0) {
    float a = 0.f, b = 0.f;
    for (int c = 0; c < CCAM; ++c)
      if (sc[c] > 0) { a += sA[c] / sc[c]; b += sB[c] / sc[c]; }
    out[0] = a;
    out[1] = 0.5f * b;   // LW
  }
}

extern "C" void kernel_launch(void* const* d_in, const int* in_sizes, int n_in,
                              void* d_out, int out_size, void* d_ws, size_t ws_size,
                              hipStream_t stream) {
  const float* feats   = (const float*)d_in[0];
  const int*   indexes = (const int*)d_in[1];
  const int*   ltab    = (const int*)d_in[2];
  const int*   ctab    = (const int*)d_in[3];
  const float* centers = (const float*)d_in[4];
  float* out = (float*)d_out;
  if (ws_size < WS_NEEDED) return;

  char* ws = (char*)d_ws;
  _Float16* cenH   = (_Float16*)(ws + OFF_CENH);
  _Float16* featsH = (_Float16*)(ws + OFF_FEATH);
  int*   labs    = (int*)(ws + OFF_LAB);
  int*   cams    = (int*)(ws + OFF_CAM);
  float* tiG     = (float*)(ws + OFF_TI);
  float* posG    = (float*)(ws + OFF_POS);
  float* partM   = (float*)(ws + OFF_PM);
  float* partS   = (float*)(ws + OFF_PS);
  float* partTop = (float*)(ws + OFF_TOP);
  float* lossIn  = (float*)(ws + OFF_LIN);
  float* lossIt  = (float*)(ws + OFF_LIT);

  cvt_centers<<<2048, 256, 0, stream>>>(centers, cenH, PPROX * DDIM / 4);
  prep_kernel<<<NSAMP, 256, 0, stream>>>(feats, indexes, ltab, ctab, featsH, labs, cams);
  gemm_stats_kernel<<<NBLK, 256, 0, stream>>>(featsH, cenH, labs, cams,
                                              posG, tiG, partM, partS, partTop);
  finalize_kernel<<<2, 256, 0, stream>>>(partM, partS, partTop, posG, tiG, lossIn, lossIt);
  reduce_kernel<<<1, 64, 0, stream>>>(lossIn, lossIt, cams, out);
}
